// LinearEmbed_33621003993929
// MI455X (gfx1250) — matrix operand order repaired and checked by reading, not compiled, not run on hardware
//
#include <hip/hip_runtime.h>

// ---------------- problem constants ----------------
#define G_    64
#define NP_   64
#define EP_   512
#define H_    128
#define IN_F_ 32
#define EDGE_F_ 16
#define L_    4
#define NTOT_ (G_ * NP_)     // 4096
#define ETOT_ (G_ * EP_)     // 32768
#define BN_EPS_ 1e-5f

#define LDA_ 136   // padded LDS stride (elements) for A tile
#define LDB_ 136   // padded LDS stride (elements) for transposed B

typedef __attribute__((ext_vector_type(16))) __bf16 v16bf;
typedef __attribute__((ext_vector_type(8)))  __bf16 v8bf;
typedef __attribute__((ext_vector_type(8)))  float  v8f;

// ---------------- WMMA fragment loaders (16-bit, 16x16x32) ----------------
// A (16x32, M=lane&15): lane-half h holds K = h*8+j (j<8) and 16+h*8+(j-8)
// lA row-major [64][LDA_]; both 8-element chunks are contiguous & 16B aligned.
__device__ __forceinline__ v16bf fragA(const __bf16* __restrict__ lA, int row, int half, int k0) {
  const __bf16* p = lA + row * LDA_ + k0 + half * 8;
  v8bf lo = *(const v8bf*)p;
  v8bf hi = *(const v8bf*)(p + 16);
  return __builtin_shufflevector(lo, hi, 0, 1, 2, 3, 4, 5, 6, 7,
                                         8, 9, 10, 11, 12, 13, 14, 15);
}
// B (32x16, N=lane&15): lane-half h holds K = h*8+j (j<8) and 16+h*8+(j-8)
// lBt transposed [128 cols][LDB_]; each 8-element K chunk is contiguous (16 B).
__device__ __forceinline__ v16bf fragB(const __bf16* __restrict__ lBt, int col, int half, int k0) {
  const __bf16* p = lBt + col * LDB_ + k0 + half * 8;
  v8bf lo = *(const v8bf*)p;
  v8bf hi = *(const v8bf*)(p + 16);
  return __builtin_shufflevector(lo, hi, 0, 1, 2, 3, 4, 5, 6, 7,
                                         8, 9, 10, 11, 12, 13, 14, 15);
}

// ---------------- generic fused GEMM: C[M,128] = act(A[M,K] @ B[K,128] + bias) ----------------
// bf16 compute, f32 accumulate. M-tile = 64 rows per block, 8 waves:
// wave w owns rows [16*(w>>1), +16) x cols [64*(w&1), +64) (4 n-tiles of 16).
__global__ __launch_bounds__(256) void gemm_h128(const float* __restrict__ A,
                                                 const float* __restrict__ B,
                                                 const float* __restrict__ bias,
                                                 float* __restrict__ C,
                                                 int M, int K, int relu) {
  __shared__ __bf16 lA[64 * LDA_];    // ~17 KB
  __shared__ __bf16 lBt[128 * LDB_];  // ~34 KB (transposed, padded)
  const int tid = threadIdx.x;
  const int Kpad = (K + 31) & ~31;
  const int m_base = blockIdx.x * 64;

  // stage B transposed (global read coalesced, LDS write strided w/ pad)
  for (int idx = tid; idx < Kpad * 128; idx += 256) {
    int r = idx >> 7, c = idx & 127;
    lBt[c * LDB_ + r] = (r < K) ? (__bf16)B[r * 128 + c] : (__bf16)0.0f;
  }
  // stage A tile (64 x Kpad)
  for (int idx = tid; idx < 64 * Kpad; idx += 256) {
    int r = idx / Kpad, c = idx - r * Kpad;
    int gr = m_base + r;
    lA[r * LDA_ + c] = (gr < M && c < K) ? (__bf16)A[(size_t)gr * K + c] : (__bf16)0.0f;
  }
  __syncthreads();

  const int wave = tid >> 5, lane = tid & 31;
  const int half = lane >> 4, l16 = lane & 15;
  const int arow = (wave >> 1) * 16 + l16;   // A row within tile
  const int nbase = (wave & 1) * 64;         // 0 or 64 -> 4 n-tiles of 16

  v8f acc[4];
#pragma unroll
  for (int nt = 0; nt < 4; ++nt)
#pragma unroll
    for (int r = 0; r < 8; ++r) acc[nt][r] = 0.0f;

  if (Kpad == 128) {
    // hot path: fully unrolled, 16 WMMAs
#pragma unroll
    for (int ks = 0; ks < 4; ++ks) {
      const int k0 = ks * 32;
      v16bf a = fragA(lA, arow, half, k0);
#pragma unroll
      for (int nt = 0; nt < 4; ++nt) {
        v16bf b = fragB(lBt, nbase + nt * 16 + l16, half, k0);
        acc[nt] = __builtin_amdgcn_wmma_f32_16x16x32_bf16(
            false, a, false, b, (short)0, acc[nt], false, false);
      }
    }
  } else {
    for (int k0 = 0; k0 < Kpad; k0 += 32) {
      v16bf a = fragA(lA, arow, half, k0);
#pragma unroll
      for (int nt = 0; nt < 4; ++nt) {
        v16bf b = fragB(lBt, nbase + nt * 16 + l16, half, k0);
        acc[nt] = __builtin_amdgcn_wmma_f32_16x16x32_bf16(
            false, a, false, b, (short)0, acc[nt], false, false);
      }
    }
  }

#pragma unroll
  for (int nt = 0; nt < 4; ++nt) {
    int gcol = nbase + nt * 16 + l16;
    float bs = bias ? bias[gcol] : 0.0f;
#pragma unroll
    for (int r = 0; r < 8; ++r) {
      int grow = m_base + (arow - l16) + half * 8 + r;  // D layout: M = half*8 + r
      if (grow < M) {
        float v = acc[nt][r] + bs;
        if (relu) v = fmaxf(v, 0.0f);
        C[(size_t)grow * 128 + gcol] = v;
      }
    }
  }
}

// ---------------- small utility kernels ----------------
__global__ void zero_f32(float* p, int n) {
  int i = blockIdx.x * 256 + threadIdx.x;
  if (i < n) p[i] = 0.0f;
}
__global__ void copy_f32(const float* __restrict__ a, float* __restrict__ b, int n) {
  int i = blockIdx.x * 256 + threadIdx.x;
  if (i < n) b[i] = a[i];
}

// messages: z[dst] += relu(h[src] + el), z preloaded with h
__global__ void scatter_msg(const float* __restrict__ h, const float* __restrict__ el,
                            const int* __restrict__ src, const int* __restrict__ dst,
                            float* __restrict__ z, int total) {
  int i = blockIdx.x * 256 + threadIdx.x;
  if (i >= total) return;
  int eidx = i >> 7, c = i & 127;
  float m = fmaxf(h[(size_t)src[eidx] * 128 + c] + el[i], 0.0f);
  unsafeAtomicAdd(&z[(size_t)dst[eidx] * 128 + c], m);
}

// batchnorm pass 1: per-column sum / sumsq (stride multiple of 128 keeps column fixed per thread)
__global__ __launch_bounds__(256) void bn_reduce(const float* __restrict__ z,
                                                 float* __restrict__ stats, int n) {
  __shared__ float ls[256], lss[256];
  float s = 0.0f, ss = 0.0f;
  for (int i = blockIdx.x * 256 + threadIdx.x; i < n; i += gridDim.x * 256) {
    float v = z[i];
    s += v; ss += v * v;
  }
  ls[threadIdx.x] = s; lss[threadIdx.x] = ss;
  __syncthreads();
  if (threadIdx.x < 128) {
    unsafeAtomicAdd(&stats[threadIdx.x], ls[threadIdx.x] + ls[threadIdx.x + 128]);
    unsafeAtomicAdd(&stats[128 + threadIdx.x], lss[threadIdx.x] + lss[threadIdx.x + 128]);
  }
}

// batchnorm pass 2 + affine + relu -> h
__global__ void bn_norm(const float* __restrict__ z, const float* __restrict__ stats,
                        const float* __restrict__ gamma, const float* __restrict__ beta,
                        float* __restrict__ hout, int n) {
  int i = blockIdx.x * 256 + threadIdx.x;
  if (i >= n) return;
  int c = i & 127;
  float mu = stats[c] * (1.0f / (float)NTOT_);
  float var = stats[128 + c] * (1.0f / (float)NTOT_) - mu * mu;
  float inv = rsqrtf(var + BN_EPS_);
  float v = (z[i] - mu) * inv * gamma[c] + beta[c];
  hout[i] = fmaxf(v, 0.0f);
}

// ---------------- final all-pairs MLP (decomposed) ----------------
// out[g,i,j] = w2 . relu(u[g,i] + v[g,j] + sumOverEdges(g,i,j){eW} + b1) + b2
__global__ __launch_bounds__(256) void pair_mlp(const float* __restrict__ u,
                                                const float* __restrict__ v,
                                                const float* __restrict__ eW,
                                                const int* __restrict__ src,
                                                const int* __restrict__ dst,
                                                const float* __restrict__ b1,
                                                const float* __restrict__ w2,
                                                const float* __restrict__ b2p,
                                                float* __restrict__ out) {
  __shared__ float s[NP_][H_];           // 32 KB sparse scatter tile for row i
  __shared__ float urow[H_], w2s[H_], b1s[H_];
  __shared__ int ssrc[EP_], sdst[EP_];   // 4 KB
  const int node = blockIdx.x;           // g*64 + i
  const int g = node >> 6, i = node & 63;
  const int tid = threadIdx.x;

  for (int idx = tid; idx < NP_ * H_; idx += 256) ((float*)s)[idx] = 0.0f;
  if (tid < 128) { urow[tid] = u[(size_t)node * 128 + tid]; w2s[tid] = w2[tid]; b1s[tid] = b1[tid]; }
  const int e0 = g * EP_;
  for (int k = tid; k < EP_; k += 256) { ssrc[k] = src[e0 + k]; sdst[k] = dst[e0 + k]; }
  __syncthreads();

  const int nbase = g << 6;
  for (int k = 0; k < EP_; ++k) {        // uniform branch across block
    if (ssrc[k] - nbase == i) {
      int dj = sdst[k] - nbase;
      if (tid < 128) s[dj][tid] += eW[(size_t)(e0 + k) * 128 + tid];
    }
  }
  __syncthreads();

  const float b2v = b2p[0];
  const int wave = tid >> 5, lane = tid & 31;
  const int c4 = lane * 4;
  const float4 u4  = *(const float4*)&urow[c4];
  const float4 w24 = *(const float4*)&w2s[c4];
  const float4 b14 = *(const float4*)&b1s[c4];
  for (int j = wave; j < NP_; j += 8) {
    float4 v4 = *(const float4*)&v[(size_t)(nbase + j) * 128 + c4];
    float4 s4 = *(const float4*)&s[j][c4];
    float acc = fmaxf(u4.x + v4.x + s4.x + b14.x, 0.0f) * w24.x
              + fmaxf(u4.y + v4.y + s4.y + b14.y, 0.0f) * w24.y
              + fmaxf(u4.z + v4.z + s4.z + b14.z, 0.0f) * w24.z
              + fmaxf(u4.w + v4.w + s4.w + b14.w, 0.0f) * w24.w;
#pragma unroll
    for (int off = 16; off; off >>= 1) acc += __shfl_xor(acc, off, 32);
    if (lane == 0) out[(size_t)node * NP_ + j] = acc + b2v;
  }
}

// ---------------- host orchestration ----------------
extern "C" void kernel_launch(void* const* d_in, const int* in_sizes, int n_in,
                              void* d_out, int out_size, void* d_ws, size_t ws_size,
                              hipStream_t stream) {
  (void)in_sizes; (void)n_in; (void)out_size; (void)ws_size;
  const float* x      = (const float*)d_in[0];
  const float* eattr  = (const float*)d_in[1];
  const int*   ei     = (const int*)  d_in[2];
  const float* atom_W = (const float*)d_in[3];
  const float* atom_b = (const float*)d_in[4];
  const float* bond_W = (const float*)d_in[5];
  const float* bond_b = (const float*)d_in[6];
  const float* gbm_W1 = (const float*)d_in[7];
  const float* gbm_b1 = (const float*)d_in[8];
  const float* gbm_W2 = (const float*)d_in[9];
  const float* gbm_b2 = (const float*)d_in[10];
  const float* gnn_W1 = (const float*)d_in[11];
  const float* gnn_b1 = (const float*)d_in[12];
  const float* gnn_W2 = (const float*)d_in[13];
  const float* gnn_b2 = (const float*)d_in[14];
  const float* bn_g   = (const float*)d_in[15];
  const float* bn_b   = (const float*)d_in[16];
  const float* mlp_W1 = (const float*)d_in[17];  // [384,128]
  const float* mlp_b1 = (const float*)d_in[18];
  const float* mlp_W2 = (const float*)d_in[19];  // [128,1]
  const float* mlp_b2 = (const float*)d_in[20];  // [1]
  const int* src = ei;
  const int* dst = ei + ETOT_;

  float* ws   = (float*)d_ws;
  float* h    = ws;                       // [4096,128]
  float* z    = h    + NTOT_ * H_;
  float* tmpN = z    + NTOT_ * H_;
  float* z2   = tmpN + NTOT_ * H_;
  float* u    = z2   + NTOT_ * H_;
  float* v    = u    + NTOT_ * H_;
  float* stats= v    + NTOT_ * H_;        // 256 (+pad)
  float* e    = stats + 512;              // [32768,128]
  float* tmpE = e    + (size_t)ETOT_ * H_;
  float* el   = tmpE + (size_t)ETOT_ * H_;
  float* eW   = tmpE;                     // reuse tmpE after layer loop

  dim3 blk(256);
  const int gN = (NTOT_ + 63) / 64;       // 64 blocks
  const int gE = (ETOT_ + 63) / 64;       // 512 blocks
  const int nElem = NTOT_ * H_;
  const int eElem = ETOT_ * H_;

  // encoders (K=32 / K=16, zero-padded to 32 inside the kernel)
  gemm_h128<<<gN, blk, 0, stream>>>(x,     atom_W, atom_b, h, NTOT_, IN_F_,   0);
  gemm_h128<<<gE, blk, 0, stream>>>(eattr, bond_W, bond_b, e, ETOT_, EDGE_F_, 0);

  for (int l = 0; l < L_; ++l) {
    // per-layer bond-encoder MLP on edges
    gemm_h128<<<gE, blk, 0, stream>>>(e,    gbm_W1 + l * H_ * H_, gbm_b1 + l * H_, tmpE, ETOT_, H_, 1);
    gemm_h128<<<gE, blk, 0, stream>>>(tmpE, gbm_W2 + l * H_ * H_, gbm_b2 + l * H_, el,   ETOT_, H_, 0);
    // z = h + scatter_add(relu(h[src] + el))
    copy_f32<<<(nElem + 255) / 256, blk, 0, stream>>>(h, z, nElem);
    scatter_msg<<<(eElem + 255) / 256, blk, 0, stream>>>(h, el, src, dst, z, eElem);
    // GINE nn MLP
    gemm_h128<<<gN, blk, 0, stream>>>(z,    gnn_W1 + l * H_ * H_, gnn_b1 + l * H_, tmpN, NTOT_, H_, 1);
    gemm_h128<<<gN, blk, 0, stream>>>(tmpN, gnn_W2 + l * H_ * H_, gnn_b2 + l * H_, z2,   NTOT_, H_, 0);
    // batchnorm + relu -> h
    zero_f32<<<1, blk, 0, stream>>>(stats, 256);
    bn_reduce<<<256, blk, 0, stream>>>(z2, stats, nElem);
    bn_norm<<<(nElem + 255) / 256, blk, 0, stream>>>(z2, stats, bn_g + l * H_, bn_b + l * H_, h, nElem);
  }

  // decomposed final MLP: W1 = [W1a; W1b; W1c] (rows 0-127, 128-255, 256-383)
  gemm_h128<<<gN, blk, 0, stream>>>(h, mlp_W1,               nullptr, u,  NTOT_, H_, 0);
  gemm_h128<<<gN, blk, 0, stream>>>(h, mlp_W1 + H_ * H_,     nullptr, v,  NTOT_, H_, 0);
  gemm_h128<<<gE, blk, 0, stream>>>(e, mlp_W1 + 2 * H_ * H_, nullptr, eW, ETOT_, H_, 0);

  pair_mlp<<<NTOT_, blk, 0, stream>>>(u, v, eW, src, dst, mlp_b1, mlp_W2, mlp_b2, (float*)d_out);
}
